// RaPNet_24154896072710
// MI455X (gfx1250) — matrix operand order repaired and checked by reading, not compiled, not run on hardware
//
#include <hip/hip_runtime.h>
#include <hip/hip_bf16.h>
#include <math.h>

typedef __attribute__((ext_vector_type(16))) _Float16 v16h;
typedef __attribute__((ext_vector_type(8)))  _Float16 v8h;
typedef __attribute__((ext_vector_type(8)))  float    v8f;

// ------------------------------------------------------------------
// Weight transform: f32 [Cout][Cin][kh][kw]  ->  f16 [tap][Cout][Cin]
// ------------------------------------------------------------------
__global__ void __launch_bounds__(256) cvt_w(const float* __restrict__ in,
                                             _Float16* __restrict__ out,
                                             int Cout, int Cin, int T) {
  int idx = blockIdx.x * 256 + threadIdx.x;
  int n = Cout * Cin * T;
  if (idx >= n) return;
  int co  = idx / (Cin * T);
  int r   = idx - co * (Cin * T);
  int ci  = r / T;
  int tap = r - ci * T;
  out[((size_t)tap * Cout + co) * Cin + ci] = (_Float16)in[idx];
}

// input image: NCHW f32 (3x256x256) -> NHWC f16
__global__ void __launch_bounds__(256) cvt_x(const float* __restrict__ x,
                                             _Float16* __restrict__ xh) {
  int idx = blockIdx.x * 256 + threadIdx.x;
  if (idx >= 3 * 65536) return;
  int c = idx >> 16, p = idx & 65535;
  xh[(size_t)p * 3 + c] = (_Float16)x[idx];
}

// ------------------------------------------------------------------
// Tap-decomposed implicit-GEMM conv via V_WMMA_F32_16X16X32_F16.
// Activations NHWC f16 -> B fragment = 2 x b128 contiguous loads.
// Block = 128 threads (4 waves): 64 pixels x 64 couts.
//   - per K-step the 64x32 f16 weight panel is staged cooperatively in LDS
//     (shared by 4 waves), fragments read back as ds b128 loads;
//   - each wave reuses its B fragment across 4 WMMAs (4 cout tiles).
// Epilogue fuses bias/BN/ReLU; writes NHWC f16 (packed b128 store) and
// optionally NCHW f32 (final feature map).
// ------------------------------------------------------------------
template<int KS, int DIL, int PAD>
__global__ void __launch_bounds__(128) conv_wmma(
    const _Float16* __restrict__ xh, const _Float16* __restrict__ wt,
    const float* __restrict__ bias,
    const float* __restrict__ bng, const float* __restrict__ bnb,
    const float* __restrict__ bnm, const float* __restrict__ bnv,
    _Float16* __restrict__ out16, float* __restrict__ out32,
    int Cin, int Cout, int OStr, int H, int W, int wshift, int doRelu)
{
  __shared__ __align__(16) _Float16 As[64 * 32];   // 4 KB weight panel

  const int lane = threadIdx.x & 31;
  const int wave = threadIdx.x >> 5;
  const int HWp  = H * W;
  const int pixBase = (blockIdx.x * 4 + wave) * 16;
  const int coG     = blockIdx.y * 64;
  const int TC      = ((Cout - coG) >= 64) ? 4 : ((Cout - coG) >> 4);

  const int hh  = lane >> 4;          // half-wave 0/1
  const int lan = lane & 15;
  const int kA0 = hh * 8;             // unused: A is read with kB0, same K-split as B
  const int kB0 = hh * 16;            // B K-split: 0-15/16-31

  const int p  = pixBase + lan;
  const int py = p >> wshift;
  const int px = p & (W - 1);

  const bool cinAligned = (Cin & 31) == 0;
  const int  T = KS * KS;

  v8f acc0 = {}, acc1 = {}, acc2 = {}, acc3 = {};

  for (int tap = 0; tap < T; ++tap) {
    const int kh = tap / KS, kw = tap - kh * KS;
    const int iy = py + kh * DIL - PAD;
    const int ix = px + kw * DIL - PAD;
    const bool valid = (iy >= 0) && (iy < H) && (ix >= 0) && (ix < W);
    const _Float16* xrow =
        valid ? (xh + (size_t)(iy * W + ix) * (size_t)Cin) : (const _Float16*)0;
    const _Float16* __restrict__ wtap = wt + (size_t)tap * Cout * Cin;

    for (int ci0 = 0; ci0 < Cin; ci0 += 32) {
      __syncthreads();                                   // LDS reuse fence
      if (cinAligned) {
        // 256 x v8h segments: 2 per thread
#pragma unroll
        for (int it = 0; it < 2; ++it) {
          int s   = threadIdx.x + it * 128;
          int row = s >> 2, seg = s & 3;
          int co  = coG + row;
          v8h vv = {};
          if (co < Cout)
            vv = *(const v8h*)(wtap + (size_t)co * Cin + ci0 + seg * 8);
          *(v8h*)(As + row * 32 + seg * 8) = vv;
        }
        // prefetch next weight panel (global_prefetch_b8)
        if (ci0 + 32 < Cin) {
          int co = coG + (threadIdx.x >> 1);
          if (co < Cout)
            __builtin_prefetch(wtap + (size_t)co * Cin + ci0 + 32, 0, 2);
        }
      } else {                                           // conv1 (Cin=3): zero-padded
        for (int s = threadIdx.x; s < 2048; s += 128) {
          int row = s >> 5, c = s & 31;
          int co = coG + row, ci = ci0 + c;
          As[s] = (co < Cout && ci < Cin) ? wtap[(size_t)co * Cin + ci]
                                          : (_Float16)0.f;
        }
      }
      __syncthreads();

      // B fragment: NHWC -> 2 contiguous b128 loads (or scalar for Cin<32)
      v16h b;
      if (cinAligned) {
        v8h blo = {}, bhi = {};
        if (valid) {
          const v8h* bp = (const v8h*)(xrow + ci0 + kB0);
          blo = bp[0];
          bhi = bp[1];
        }
        b = __builtin_shufflevector(blo, bhi, 0, 1, 2, 3, 4, 5, 6, 7,
                                              8, 9, 10, 11, 12, 13, 14, 15);
      } else {
#pragma unroll
        for (int idx = 0; idx < 16; ++idx) {
          int ci = ci0 + kB0 + idx;
          b[idx] = (valid && ci < Cin) ? xrow[ci] : (_Float16)0.f;
        }
      }

      // 4 cout tiles share this B fragment
#pragma unroll
      for (int t = 0; t < 4; ++t) {
        if (t < TC) {
          const _Float16* arow = As + (t * 16 + lan) * 32;
          v8h lo = *(const v8h*)(arow + kB0);            // K = kB0..kB0+7
          v8h hi = *(const v8h*)(arow + kB0 + 8);        // K = kB0+8..+15
          v16h a = __builtin_shufflevector(lo, hi, 0, 1, 2, 3, 4, 5, 6, 7,
                                                   8, 9, 10, 11, 12, 13, 14, 15);
          v8f* pa = (t == 0) ? &acc0 : (t == 1) ? &acc1 : (t == 2) ? &acc2 : &acc3;
          *pa = __builtin_amdgcn_wmma_f32_16x16x32_f16(false, a, false, b,
                                                       (short)0, *pa, false, false);
        }
      }
    }
  }

  // Epilogue. C/D layout: lanes 0-15 VGPR j -> M=j; lanes 16-31 -> M=8+j.
#pragma unroll
  for (int t = 0; t < 4; ++t) {
    if (t < TC) {
      v8f accv = (t == 0) ? acc0 : (t == 1) ? acc1 : (t == 2) ? acc2 : acc3;
      const int co0 = coG + t * 16 + hh * 8;
      float fv[8];
      v8h ov;
#pragma unroll
      for (int j = 0; j < 8; ++j) {
        int co = co0 + j;
        float v = accv[j];
        if (bias) v += bias[co];
        if (bng)  v = bng[co] * (v - bnm[co]) * rsqrtf(bnv[co] + 1e-5f) + bnb[co];
        if (doRelu) v = fmaxf(v, 0.f);
        fv[j] = v;
        ov[j] = (_Float16)v;
      }
      *(v8h*)(out16 + (size_t)p * OStr + co0) = ov;      // packed NHWC store
      if (out32) {
#pragma unroll
        for (int j = 0; j < 8; ++j)
          out32[(size_t)(co0 + j) * HWp + p] = fv[j];    // NCHW f32 (final f)
      }
    }
  }
}

// ------------------------------------------------------------------
// NHWC f16 pooling (maxpool: -inf init, skip OOB; avgpool: 0-pad, /k^2)
// channel-innermost -> fully coalesced
// ------------------------------------------------------------------
__global__ void __launch_bounds__(256) pool_nhwc(
    const _Float16* __restrict__ in, _Float16* __restrict__ out,
    int C, int H, int W, int Ho, int Wo, int k, int s, int p, int isMax)
{
  size_t idx = (size_t)blockIdx.x * 256 + threadIdx.x;
  size_t total = (size_t)C * Ho * Wo;
  if (idx >= total) return;
  int c = (int)(idx % C);
  size_t t = idx / C;
  int ox = (int)(t % Wo);
  int oy = (int)(t / Wo);
  float acc = isMax ? -__builtin_inff() : 0.f;
  for (int dy = 0; dy < k; ++dy) {
    int iy = oy * s - p + dy;
    if (iy < 0 || iy >= H) continue;
    for (int dx = 0; dx < k; ++dx) {
      int ix = ox * s - p + dx;
      if (ix < 0 || ix >= W) continue;
      float v = (float)in[((size_t)iy * W + ix) * C + c];
      acc = isMax ? fmaxf(acc, v) : (acc + v);
    }
  }
  out[((size_t)oy * Wo + ox) * C + c] =
      (_Float16)(isMax ? acc : acc / (float)(k * k));
}

// ------------------------------------------------------------------
// Attention head: 1x1 conv (96->1) + softplus + 2x nearest upsample
// fusion is NHWC f16 (96 channels contiguous per pixel)
// ------------------------------------------------------------------
__global__ void __launch_bounds__(256) att_kernel(
    const _Float16* __restrict__ fusion, const float* __restrict__ attw,
    const float* __restrict__ attb, float* __restrict__ att)
{
  int idx = blockIdx.x * 256 + threadIdx.x;      // 256*256 outputs
  if (idx >= 65536) return;
  int x = idx & 255, y = idx >> 8;
  size_t sp = (size_t)((y >> 1) * 128 + (x >> 1)) * 96;
  float s = attb[0];
  for (int c = 0; c < 96; ++c) s += (float)fusion[sp + c] * attw[c];
  att[idx] = (s > 20.f) ? s : log1pf(expf(s));   // stable softplus
}

// ------------------------------------------------------------------
// Post-processing (operates on NCHW f32 feature map f)
// ------------------------------------------------------------------
__global__ void __launch_bounds__(256) chanmax_kernel(
    const float* __restrict__ f, float* __restrict__ cm)
{
  int p = blockIdx.x * 256 + threadIdx.x;
  if (p >= 65536) return;
  float m = -__builtin_inff();
  for (int c = 0; c < 512; ++c) m = fmaxf(m, f[((size_t)c << 16) + p]);
  cm[p] = m;
}

// horizontal 27-window max (separable half of 27x27 maxpool, pad 13)
__global__ void __launch_bounds__(256) hmax_kernel(
    const float* __restrict__ f, float* __restrict__ tmp)
{
  size_t idx = (size_t)blockIdx.x * 256 + threadIdx.x;
  if (idx >= (size_t)512 * 65536) return;
  int p = (int)(idx & 65535);
  int x = p & 255, y = p >> 8;
  const float* row = f + (idx >> 16 << 16) + (size_t)(y << 8);
  float m = -__builtin_inff();
  for (int dx = -13; dx <= 13; ++dx) {
    int ix = x + dx;
    if (ix >= 0 && ix < 256) m = fmaxf(m, row[ix]);
  }
  tmp[idx] = m;
}

__device__ inline float at_pad(const float* fc, int y, int x) {
  return (y >= 0 && y < 256 && x >= 0 && x < 256) ? fc[(y << 8) + x] : 0.f;
}

__device__ inline void derivs(const float* __restrict__ fc, int y, int x,
                              float& di, float& dj, float& dii, float& dij, float& djj)
{
  float u  = at_pad(fc, y - 1, x),     d  = at_pad(fc, y + 1, x);
  float l  = at_pad(fc, y, x - 1),     r  = at_pad(fc, y, x + 1);
  float ul = at_pad(fc, y - 1, x - 1), ur = at_pad(fc, y - 1, x + 1);
  float dl = at_pad(fc, y + 1, x - 1), dr = at_pad(fc, y + 1, x + 1);
  float c  = fc[(y << 8) + x];
  di  = 0.5f * (d - u);
  dj  = 0.5f * (r - l);
  dii = u - 2.f * c + d;
  djj = l - 2.f * c + r;
  dij = 0.25f * (ul - ur - dl + dr);
}

__global__ void __launch_bounds__(256) detect_kernel(
    const float* __restrict__ f, const float* __restrict__ tmp,
    const float* __restrict__ cm, float* __restrict__ det)
{
  size_t idx = (size_t)blockIdx.x * 256 + threadIdx.x;
  if (idx >= (size_t)512 * 65536) return;
  int p = (int)(idx & 65535);
  int x = p & 255, y = p >> 8;
  const float* fc = f + (idx >> 16 << 16);
  const float* tc = tmp + (idx >> 16 << 16);
  float v = fc[p];
  float lm = -__builtin_inff();                 // vertical half of 27x27 max
  for (int dy = -13; dy <= 13; ++dy) {
    int iy = y + dy;
    if (iy >= 0 && iy < 256) lm = fmaxf(lm, tc[(iy << 8) + x]);
  }
  float di, dj, dii, dij, djj;
  derivs(fc, y, x, di, dj, dii, dij, djj);
  float dH = dii * djj - dij * dij;
  float tr = dii + djj;
  bool notedge = (tr * tr / dH <= 7.2f) && (dH > 0.f);   // (5+1)^2/5
  bool hit = (v == cm[p]) && (v == lm) && notedge;
  det[idx] = hit ? 1.f : 0.f;
}

__global__ void __launch_bounds__(256) disp_kernel(
    const float* __restrict__ f, float* __restrict__ disp)
{
  size_t idx = (size_t)blockIdx.x * 256 + threadIdx.x;
  const size_t FN = (size_t)512 * 65536;
  if (idx >= FN) return;
  int p = (int)(idx & 65535);
  int x = p & 255, y = p >> 8;
  const float* fc = f + (idx >> 16 << 16);
  float di, dj, dii, dij, djj;
  derivs(fc, y, x, di, dj, dii, dij, djj);
  float dH = dii * djj - dij * dij;
  float inv00 = djj / dH, inv01 = -dij / dH, inv11 = dii / dH;
  disp[idx]      = -(inv00 * di + inv01 * dj);   // step_i
  disp[FN + idx] = -(inv01 * di + inv11 * dj);   // step_j
}

// ------------------------------------------------------------------
// Host-side launch helpers
// ------------------------------------------------------------------
template<int KS, int DIL, int PAD>
static void launch_conv(const _Float16* x, const _Float16* w, const float* bias,
                        const float* g, const float* be, const float* m, const float* v,
                        _Float16* out16, float* out32, int Cin, int Cout, int OStr,
                        int H, int W, int wshift, int relu, hipStream_t s)
{
  dim3 grid((unsigned)((H * W) / 64), (unsigned)((Cout + 63) / 64));
  conv_wmma<KS, DIL, PAD><<<grid, dim3(128), 0, s>>>(
      x, w, bias, g, be, m, v, out16, out32, Cin, Cout, OStr, H, W, wshift, relu);
}

static void launch_pool(const _Float16* in, _Float16* out, int C, int H, int W,
                        int Ho, int Wo, int k, int s, int p, int isMax, hipStream_t st)
{
  size_t total = (size_t)C * Ho * Wo;
  pool_nhwc<<<dim3((unsigned)((total + 255) / 256)), dim3(256), 0, st>>>(
      in, out, C, H, W, Ho, Wo, k, s, p, isMax);
}

extern "C" void kernel_launch(void* const* d_in, const int* in_sizes, int n_in,
                              void* d_out, int out_size, void* d_ws, size_t ws_size,
                              hipStream_t stream)
{
  (void)in_sizes; (void)n_in; (void)out_size; (void)ws_size;

  const float* x = (const float*)d_in[0];
  const float *W[10], *B[10];
  for (int i = 0; i < 10; ++i) { W[i] = (const float*)d_in[1 + 2 * i];
                                 B[i] = (const float*)d_in[2 + 2 * i]; }
  const float *AW[3], *G[3], *BE[3], *M[3], *V[3];
  for (int j = 0; j < 3; ++j) {
    AW[j] = (const float*)d_in[21 + 5 * j];
    G[j]  = (const float*)d_in[22 + 5 * j];
    BE[j] = (const float*)d_in[23 + 5 * j];
    M[j]  = (const float*)d_in[24 + 5 * j];
    V[j]  = (const float*)d_in[25 + 5 * j];
  }
  const float* attw = (const float*)d_in[36];
  const float* attb = (const float*)d_in[37];

  // d_out layout: [f 512x256x256][att 256x256][det 512x256x256][disp 2x512x256x256]
  const size_t FN = (size_t)512 * 65536;
  float* out     = (float*)d_out;
  float* fOut    = out;
  float* attOut  = out + FN;
  float* detOut  = attOut + 65536;
  float* dispOut = detOut + FN;

  // NHWC f16 ping-pong buffers (64 MB each) both inside the det region (134 MB)
  _Float16* hA = (_Float16*)detOut;
  _Float16* hB = hA + FN;
  // attention f16 scratch + converted input live in disp[0] (consumed before
  // post-processing overwrites it); chanmax lives in disp[1]
  _Float16* x16   = (_Float16*)dispOut;                  // 3*65536 halves
  _Float16* mp16  = (_Float16*)dispOut;                  // 512*128*128 (reused after x16 dead)
  _Float16* aB16  = mp16 + (size_t)512 * 16384;
  _Float16* fus16 = aB16 + (size_t)512 * 16384;          // 96*128*128
  float*    cmax  = dispOut + FN;                        // disp[1] start

  // ---- stage transposed f16 weights in d_ws ----
  static const int ci10[10] = {3, 64, 64, 128, 128, 256, 256, 256, 512, 512};
  static const int co10[10] = {64, 64, 128, 128, 256, 256, 256, 512, 512, 512};
  static const int aks[3]   = {3, 5, 7};
  _Float16* wh[10]; _Float16* awh[3];
  {
    _Float16* cur = (_Float16*)d_ws;
    for (int i = 0; i < 10; ++i) {
      int n = co10[i] * ci10[i] * 9;
      wh[i] = cur; cur += n;
      cvt_w<<<dim3((n + 255) / 256), dim3(256), 0, stream>>>(W[i], wh[i],
                                                             co10[i], ci10[i], 9);
    }
    for (int j = 0; j < 3; ++j) {
      int T = aks[j] * aks[j];
      int n = 32 * 512 * T;
      awh[j] = cur; cur += n;
      cvt_w<<<dim3((n + 255) / 256), dim3(256), 0, stream>>>(AW[j], awh[j],
                                                             32, 512, T);
    }
  }
  cvt_x<<<dim3(768), dim3(256), 0, stream>>>(x, x16);

  // ---- backbone (256x256, wshift=8), NHWC f16 activations ----
  launch_conv<3,1,1>(x16, wh[0], B[0], 0,0,0,0, hA, 0,   3,  64,  64, 256,256,8, 1, stream);
  launch_conv<3,1,1>(hA,  wh[1], B[1], 0,0,0,0, hB, 0,  64,  64,  64, 256,256,8, 1, stream);
  launch_pool(hB, hA, 64, 256,256, 256,256, 3,1,1, 1, stream);
  launch_conv<3,1,1>(hA,  wh[2], B[2], 0,0,0,0, hB, 0,  64, 128, 128, 256,256,8, 1, stream);
  launch_conv<3,1,1>(hB,  wh[3], B[3], 0,0,0,0, hA, 0, 128, 128, 128, 256,256,8, 1, stream);
  launch_pool(hA, hB, 128, 256,256, 256,256, 3,1,1, 1, stream);
  launch_conv<3,1,1>(hB,  wh[4], B[4], 0,0,0,0, hA, 0, 128, 256, 256, 256,256,8, 1, stream);
  launch_conv<3,1,1>(hA,  wh[5], B[5], 0,0,0,0, hB, 0, 256, 256, 256, 256,256,8, 1, stream);
  launch_conv<3,1,1>(hB,  wh[6], B[6], 0,0,0,0, hA, 0, 256, 256, 256, 256,256,8, 1, stream);
  launch_pool(hA, hB, 256, 256,256, 256,256, 3,1,1, 0, stream);   // avgpool
  launch_conv<3,2,2>(hB,  wh[7], B[7], 0,0,0,0, hA, 0, 256, 512, 512, 256,256,8, 1, stream);
  launch_conv<3,2,2>(hA,  wh[8], B[8], 0,0,0,0, hB, 0, 512, 512, 512, 256,256,8, 1, stream);
  launch_conv<3,2,2>(hB,  wh[9], B[9], 0,0,0,0, hA, fOut,
                     512, 512, 512, 256,256,8, 1, stream);        // f: NHWC f16 + NCHW f32

  // ---- attention head (128x128, wshift=7), NHWC f16 ----
  launch_pool(hA,   mp16, 512, 256,256, 128,128, 3,2,1, 1, stream);  // maxpool s2
  launch_pool(mp16, aB16, 512, 128,128, 128,128, 3,1,1, 0, stream);  // avgpool
  launch_conv<3,1,1>(aB16, awh[0], 0, G[0],BE[0],M[0],V[0], fus16 +  0, 0, 512, 32, 96, 128,128,7, 1, stream);
  launch_conv<5,1,2>(aB16, awh[1], 0, G[1],BE[1],M[1],V[1], fus16 + 32, 0, 512, 32, 96, 128,128,7, 1, stream);
  launch_conv<7,1,3>(aB16, awh[2], 0, G[2],BE[2],M[2],V[2], fus16 + 64, 0, 512, 32, 96, 128,128,7, 1, stream);
  att_kernel<<<dim3(256), dim3(256), 0, stream>>>(fus16, attw, attb, attOut);

  // ---- detections + displacements (NCHW f32 f) ----
  const unsigned gFull = (unsigned)(FN / 256);
  chanmax_kernel<<<dim3(256), dim3(256), 0, stream>>>(fOut, cmax);
  hmax_kernel  <<<dim3(gFull), dim3(256), 0, stream>>>(fOut, dispOut);       // row-max temp in disp[0]
  detect_kernel<<<dim3(gFull), dim3(256), 0, stream>>>(fOut, dispOut, cmax, detOut);
  disp_kernel  <<<dim3(gFull), dim3(256), 0, stream>>>(fOut, dispOut);       // overwrites temp with final
}
